// CIN_39333310497372
// MI455X (gfx1250) — matrix operand order repaired and checked by reading, not compiled, not run on hardware
//
#include <hip/hip_runtime.h>
#include <hip/hip_bf16.h>

typedef __attribute__((ext_vector_type(16))) __bf16 v16bf;
typedef __attribute__((ext_vector_type(8)))  __bf16 v8bf;
typedef __attribute__((ext_vector_type(8)))  float  v8f;
typedef __attribute__((ext_vector_type(4)))  float  v4f;

#define B_TOT  4096
#define M_DIM  40
#define D_DIM  16
#define O_DIM  128
#define NB     4      // batch elements per workgroup
#define GROUP  5      // K-chunks per phase; divides 50 and 160 -> no tail, full unroll

#define X0_PITCH 44   // x0t[d][m] row pitch: 16B-aligned v4f loads, conflict-free
#define XT_PITCH 129  // xt[d][h] row pitch: stride 129 = conflict-free scalar access

#define HM0   (M_DIM * M_DIM)    // 1600
#define NC0   (HM0 / 32)         // 50 chunks
#define HM12  (O_DIM * M_DIM)    // 5120
#define NC12  (HM12 / 32)        // 160 chunks
#define PACK0  (8 * NC0 * 512)
#define PACK12 (8 * NC12 * 512)
#define PACK_TOTAL (PACK0 + 2 * PACK12)   // ~3.0 MB bf16 scratch

// ---------------------------------------------------------------------------
// Prep: fp32 kernels -> bf16, pre-swizzled into WMMA A-fragment order.
// Layout per layer: [otile(8)][chunk(nc)][lane(32)][j(16)] halves.
// A-fragment (16x32 bf16): lane l = row M (l&15); component j holds
// K = (j>>3)*16 + (l>>4)*8 + (j&7).
// ---------------------------------------------------------------------------
__global__ __launch_bounds__(256) void cin_pack_k(const float* __restrict__ k0,
                                                  const float* __restrict__ k1,
                                                  const float* __restrict__ k2,
                                                  __bf16* __restrict__ ws) {
    int idx = blockIdx.x * 256 + threadIdx.x;
    if (idx >= PACK_TOTAL) return;
    const float* src;
    __bf16* dst;
    int HM, local;
    if (idx < PACK0)               { src = k0; HM = HM0;  local = idx;                  dst = ws; }
    else if (idx < PACK0 + PACK12) { src = k1; HM = HM12; local = idx - PACK0;          dst = ws + PACK0; }
    else                           { src = k2; HM = HM12; local = idx - PACK0 - PACK12; dst = ws + PACK0 + PACK12; }
    const int nc    = HM / 32;
    const int otile = local / (nc * 512);
    const int rem   = local % (nc * 512);
    const int chunk = rem / 512;
    const int e     = rem % 512;
    const int lane  = e >> 4;
    const int j     = e & 15;
    const int o     = otile * 16 + (lane & 15);
    const int g     = lane >> 4;
    const int kloc  = ((j >> 3) << 4) + g * 8 + (j & 7);
    const int hm    = chunk * 32 + kloc;
    dst[local] = (__bf16)src[(size_t)o * HM + hm];
}

// ---------------------------------------------------------------------------
// Main kernel: 256 threads = 8 wave32, NB=4 batch elements per workgroup.
// Wave w owns o-tile [16w,16w+16). Hidden state kept transposed xt[d][h] in
// LDS (fp32). Build phase writes C in B-fragment order with b128 stores.
// ---------------------------------------------------------------------------
__global__ __launch_bounds__(256) void cin_main(const float* __restrict__ x0g,
                                                const __bf16* __restrict__ kpack,
                                                float* __restrict__ out) {
    __shared__ float s_x0t[NB * D_DIM * X0_PITCH];            // 11.0 KB : x0t[bl][d][m]
    __shared__ float s_xt [NB * D_DIM * XT_PITCH];            // 32.3 KB : xt [bl][d][h]
    __shared__ __align__(32) __bf16 s_c[NB][GROUP * 512];     // 20.0 KB : B-fragment staging

    const int t    = threadIdx.x;
    const int lane = t & 31;
    const int w    = t >> 5;
    const int b0   = blockIdx.x * NB;

    // build-phase thread coordinates (all decompositions by compile-time constants)
    const int bbl = t >> 6;          // batch element 0..3 (64 threads each)
    const int r   = t & 63;
    const int bd  = r & 15;          // d column
    const int bg  = (r >> 4) & 1;    // K half-group within chunk
    const int bjs = r >> 5;          // j sub-run (0 -> j 0..7, 1 -> j 8..15)
    const float* x0row = &s_x0t[(bbl * D_DIM + bd) * X0_PITCH];   // this thread's x0t row
    __bf16* crow = &s_c[bbl][((bg << 4) + bd) * 16 + (bjs << 3)]; // this thread's C slot

    // stage x0 transposed: s_x0t[bl][d][m] = x0[b0+bl][m][d]  (coalesced reads)
    for (int idx = t; idx < NB * M_DIM * D_DIM; idx += 256) {
        int bl = idx / (M_DIM * D_DIM);
        int rm = idx - bl * (M_DIM * D_DIM);
        int m  = rm >> 4;
        int d  = rm & 15;
        s_x0t[(bl * D_DIM + d) * X0_PITCH + m] = x0g[(size_t)(b0 + bl) * (M_DIM * D_DIM) + rm];
    }
    __syncthreads();

    const __bf16* kp = kpack;
    const float* xcur = s_x0t;       // layer 0: x == x0 (transposed view)
    int xpitch = X0_PITCH;
    int H = M_DIM;

    for (int layer = 0; layer < 3; ++layer) {
        const int nc = (H * M_DIM) / 32;
        const __bf16* wlane = kp + (size_t)w * nc * 512 + lane * 16;  // wave's A stream
        v8f acc[NB] = {};

        for (int gstart = 0; gstart < nc; gstart += GROUP) {
            // ---- build phase: per chunk, one x[h,d] scalar + 8 contiguous
            // x0t[d][m0..m0+8) (two b128 loads) -> one b128 store of 8 bf16.
#pragma unroll
            for (int c = 0; c < GROUP; ++c) {
                const int q  = ((gstart + c) << 2) + (bjs << 1) + bg;  // hm0/8
                const int h  = q / 5;                                  // constant divide
                const int m0 = (q - h * 5) << 3;
                const float xv = xcur[(bbl * D_DIM + bd) * xpitch + h];
                v4f a0 = *(const v4f*)(x0row + m0);
                v4f a1 = *(const v4f*)(x0row + m0 + 4);
                v8bf pk;
#pragma unroll
                for (int jj = 0; jj < 4; ++jj) pk[jj]     = (__bf16)(xv * a0[jj]);
#pragma unroll
                for (int jj = 0; jj < 4; ++jj) pk[jj + 4] = (__bf16)(xv * a1[jj]);
                *(v8bf*)(crow + c * 512) = pk;
            }
            __syncthreads();

            // ---- compute phase: A from L2 (immediate-offset clause), four B
            // fragments in distinct registers so DS waits can be partial.
#pragma unroll
            for (int c = 0; c < GROUP; ++c) {
                v16bf a  = *(const v16bf*)(wlane + (size_t)(gstart + c) * 512);
                if (gstart + GROUP < nc)
                    __builtin_prefetch(wlane + (size_t)(gstart + GROUP + c) * 512, 0, 3);
                v16bf b0 = *(const v16bf*)&s_c[0][c * 512 + lane * 16];
                v16bf b1 = *(const v16bf*)&s_c[1][c * 512 + lane * 16];
                v16bf b2 = *(const v16bf*)&s_c[2][c * 512 + lane * 16];
                v16bf b3 = *(const v16bf*)&s_c[3][c * 512 + lane * 16];
                acc[0] = __builtin_amdgcn_wmma_f32_16x16x32_bf16(false, a, false, b0, (short)0, acc[0], false, false);
                acc[1] = __builtin_amdgcn_wmma_f32_16x16x32_bf16(false, a, false, b1, (short)0, acc[1], false, false);
                acc[2] = __builtin_amdgcn_wmma_f32_16x16x32_bf16(false, a, false, b2, (short)0, acc[2], false, false);
                acc[3] = __builtin_amdgcn_wmma_f32_16x16x32_bf16(false, a, false, b3, (short)0, acc[3], false, false);
            }
            __syncthreads();
        }

        // ---- write D (fp32) into transposed hidden state: xt[d][o]
        {
            const int rowbase = 16 * w + 8 * (lane >> 4);
            const int d       = lane & 15;
#pragma unroll
            for (int bl = 0; bl < NB; ++bl) {
                float* xp = &s_xt[(bl * D_DIM + d) * XT_PITCH + rowbase];
#pragma unroll
                for (int rr = 0; rr < 8; ++rr) xp[rr] = acc[bl][rr];
            }
        }
        __syncthreads();

        // ---- per-layer output: sum over d
        for (int idx = t; idx < NB * O_DIM; idx += 256) {
            int bl = idx >> 7;
            int o  = idx & (O_DIM - 1);
            float s = 0.f;
#pragma unroll
            for (int d = 0; d < D_DIM; ++d) s += s_xt[(bl * D_DIM + d) * XT_PITCH + o];
            out[(size_t)(b0 + bl) * (3 * O_DIM) + layer * O_DIM + o] = s;
        }

        kp    += (layer == 0) ? PACK0 : PACK12;
        xcur   = s_xt;
        xpitch = XT_PITCH;
        H      = O_DIM;
        __syncthreads();
    }
}

extern "C" void kernel_launch(void* const* d_in, const int* in_sizes, int n_in,
                              void* d_out, int out_size, void* d_ws, size_t ws_size,
                              hipStream_t stream) {
    (void)in_sizes; (void)n_in; (void)out_size; (void)ws_size;
    const float* x0 = (const float*)d_in[0];
    const float* k0 = (const float*)d_in[1];
    const float* k1 = (const float*)d_in[2];
    const float* k2 = (const float*)d_in[3];
    __bf16* ws = (__bf16*)d_ws;          // needs PACK_TOTAL*2 ≈ 3.0 MB scratch
    float* out = (float*)d_out;

    cin_pack_k<<<(PACK_TOTAL + 255) / 256, 256, 0, stream>>>(k0, k1, k2, ws);
    cin_main<<<B_TOT / NB, 256, 0, stream>>>(x0, ws, out);
}
